// MixtureOfExperts_20229295964739
// MI455X (gfx1250) — matrix operand order repaired and checked by reading, not compiled, not run on hardware
//
#include <hip/hip_runtime.h>
#include <math.h>

typedef __attribute__((ext_vector_type(16))) __bf16 v16bf;
typedef __attribute__((ext_vector_type(8)))  __bf16 v8bf;
typedef __attribute__((ext_vector_type(4)))  __bf16 v4bf;
typedef __attribute__((ext_vector_type(8)))  float  v8f;

namespace {
constexpr int kBn = 2, kSn = 2048, kT = kBn * kSn;   // 4096 tokens
constexpr int kD = 1024, kH = 4096, kO = 1024, kE = 8;
constexpr int kCap = kT;           // per-expert slot capacity (worst case)
constexpr int kMT  = kCap / 128;   // 32 m-tiles per expert
constexpr int kLds = 40;           // LDS tile row stride in halves (80B: conflict-free, 16B aligned)
constexpr float kEps = 1e-5f;

// ---- workspace layout (bytes) ----
constexpr size_t OFF_XB  = 0;                               // [kT][kD] bf16
constexpr size_t SZ_XB   = (size_t)kT * kD * 2;
constexpr size_t OFF_HB  = OFF_XB + SZ_XB;                  // [2*kT][kH] bf16
constexpr size_t SZ_HB   = (size_t)kT * 2 * kH * 2;
constexpr size_t OFF_YB  = OFF_HB + SZ_HB;                  // [2*kT][kO] f32
constexpr size_t SZ_YB   = (size_t)kT * 2 * kO * 4;
constexpr size_t OFF_WT  = OFF_YB + SZ_YB;                  // [2*kT] f32
constexpr size_t SZ_WT   = (size_t)kT * 2 * 4;
constexpr size_t OFF_CNT = OFF_WT + SZ_WT;                  // [kE] int (+pad)
constexpr size_t OFF_LST = OFF_CNT + 256;                   // [kE][kCap] int
} // namespace

// ---- CDNA5 async global->LDS copy (ASYNCcnt path), 32 bytes per lane ----
__device__ __forceinline__ void async_copy_32B(const __bf16* g, __bf16* l) {
    uint32_t la = (uint32_t)(uintptr_t)l;          // low 32 bits = LDS offset
    uint64_t ga = (uint64_t)(uintptr_t)g;
    asm volatile(
        "global_load_async_to_lds_b128 %0, %1, off\n\t"
        "global_load_async_to_lds_b128 %0, %1, off offset:16"
        :: "v"(la), "v"(ga) : "memory");
}
__device__ __forceinline__ void wait_async() {
    asm volatile("s_wait_asynccnt 0x0" ::: "memory");
}

// Branch-free tanh-form GELU: v_exp_f32 + v_rcp_f32, no divergence.
__device__ __forceinline__ float gelu_fast(float v) {
    float u = v * (0.7978845608028654f + 0.0356774081f * v * v);
    float a = __builtin_fabsf(u);
    float em = __expf(-2.0f * a);
    float th = (1.0f - em) * __builtin_amdgcn_rcpf(1.0f + em);
    th = __builtin_copysignf(th, u);
    return 0.5f * v * (1.0f + th);
}

__global__ void moe_init(int* __restrict__ counts) {
    if (threadIdx.x < kE) counts[threadIdx.x] = 0;
}

__global__ void moe_cvt_x(const float* __restrict__ x, __bf16* __restrict__ xb) {
    size_t i = ((size_t)blockIdx.x * blockDim.x + threadIdx.x) * 4;
    float4 v = *(const float4*)(x + i);
    v4bf o;
    o[0] = (__bf16)v.x; o[1] = (__bf16)v.y; o[2] = (__bf16)v.z; o[3] = (__bf16)v.w;
    *(v4bf*)(xb + i) = o;
}

// One wave32 per token: logits, top-2 softmax renorm, build per-expert lists.
__global__ void moe_router(const float* __restrict__ x, const float* __restrict__ Wr,
                           const float* __restrict__ br, int* __restrict__ counts,
                           int* __restrict__ tlist, float* __restrict__ wtok) {
    const int wid = threadIdx.x >> 5, lane = threadIdx.x & 31;
    const int t = blockIdx.x * 8 + wid;
    float acc[kE];
#pragma unroll
    for (int e = 0; e < kE; ++e) acc[e] = 0.f;
    const float* xr = x + (size_t)t * kD;
    for (int d = lane; d < kD; d += 32) {
        float xv = xr[d];
        const float* wr = Wr + (size_t)d * kE;
#pragma unroll
        for (int e = 0; e < kE; ++e) acc[e] += xv * wr[e];
    }
#pragma unroll
    for (int e = 0; e < kE; ++e) {
#pragma unroll
        for (int off = 16; off; off >>= 1) acc[e] += __shfl_xor(acc[e], off, 32);
    }
    if (lane == 0) {
        float l[kE];
#pragma unroll
        for (int e = 0; e < kE; ++e) l[e] = acc[e] + br[e];
        int i0 = 0; float m0v = l[0];
#pragma unroll
        for (int e = 1; e < kE; ++e) if (l[e] > m0v) { m0v = l[e]; i0 = e; }
        int i1 = (i0 == 0) ? 1 : 0; float m1v = l[i1];
#pragma unroll
        for (int e = 0; e < kE; ++e)
            if (e != i0 && l[e] > m1v) { m1v = l[e]; i1 = e; }
        float p1 = __expf(m1v - m0v);           // renormalized top-2 softmax
        float w0 = 1.f / (1.f + p1);
        wtok[2 * t]     = w0;
        wtok[2 * t + 1] = 1.f - w0;
        int s0 = atomicAdd(&counts[i0], 1);
        tlist[i0 * kCap + s0] = 2 * t;
        int s1 = atomicAdd(&counts[i1], 1);
        tlist[i1 * kCap + s1] = 2 * t + 1;
    }
}

// ---- fragment loads from LDS tiles: sA [row][k], sB [col][k], stride kLds ----
__device__ __forceinline__ void load_afrag(const __bf16* sA, int row, int lhi, v16bf& a) {
    const __bf16* p = sA + row * kLds + lhi * 8;        // k = lhi*8 + c (c<8), +16 (c>=8)
    v8bf lo = *(const v8bf*)p;
    v8bf hi = *(const v8bf*)(p + 16);
#pragma unroll
    for (int c = 0; c < 8; ++c) { a[c] = lo[c]; a[c + 8] = hi[c]; }
}
__device__ __forceinline__ void load_bfrag(const __bf16* sB, int col, int lhi, v16bf& b) {
    const __bf16* p = sB + col * kLds + lhi * 8;        // k = lhi*8 + c (c<8), +16 (c>=8)
    v8bf lo = *(const v8bf*)p;
    v8bf hi = *(const v8bf*)(p + 16);
#pragma unroll
    for (int c = 0; c < 8; ++c) { b[c] = lo[c]; b[c + 8] = hi[c]; }
}

// Double-buffered 128x128x(K) bf16 WMMA GEMM body, shared by both expert GEMMs.
// A rows are gathered (compacted slot list); B is f32 weights converted on the fly.
#define MOE_GEMM_BODY(ASRC_EXPR, LDA, LDB, KDIM)                                         \
    __shared__ __align__(16) __bf16 sA[2][128 * kLds];                                   \
    __shared__ __align__(16) __bf16 sB[2][128 * kLds];                                   \
    __shared__ int rowid[128];                                                           \
    const int tid = threadIdx.x;                                                         \
    if (tid < 128) {                                                                     \
        int s = m0 + tid;                                                                \
        rowid[tid] = tlist[e * kCap + (s < cnt ? s : cnt - 1)];                          \
    }                                                                                    \
    __syncthreads();                                                                     \
    /* A copy: 2 threads per row, 32B each, via async global->LDS */                     \
    const int ar = tid >> 1, ah = (tid & 1) * 16;                                        \
    const int arid = rowid[ar];                                                          \
    const __bf16* aSrc = (ASRC_EXPR) + ah;                                               \
    __bf16* aDst[2] = { &sA[0][ar * kLds + ah], &sA[1][ar * kLds + ah] };                \
    /* B copy: 4 consecutive n x 4 consecutive k per thread */                           \
    const int n4 = (tid & 31) * 4, kq = tid >> 5;                                        \
    const float* bSrc = Wp + (size_t)(kq * 4) * (LDB) + n0 + n4;                         \
    float wreg[4][4];                                                                    \
    const int wid = tid >> 5, lane = tid & 31;                                           \
    const int wm = (wid & 3) * 32, wn = (wid >> 2) * 64;                                 \
    const int lhi = lane >> 4, l16 = lane & 15;                                          \
    v8f acc[2][4];                                                                       \
    _Pragma("unroll") for (int i = 0; i < 2; ++i)                                        \
        _Pragma("unroll") for (int j = 0; j < 4; ++j) acc[i][j] = (v8f)0.f;              \
    /* prologue: stage tile 0 into buffer 0 */                                           \
    async_copy_32B(aSrc, aDst[0]);                                                       \
    _Pragma("unroll") for (int kk = 0; kk < 4; ++kk) {                                   \
        float4 q = *(const float4*)(bSrc + (size_t)kk * (LDB));                          \
        wreg[kk][0] = q.x; wreg[kk][1] = q.y; wreg[kk][2] = q.z; wreg[kk][3] = q.w;      \
    }                                                                                    \
    _Pragma("unroll") for (int j = 0; j < 4; ++j) {                                      \
        v4bf pk;                                                                         \
        _Pragma("unroll") for (int kk = 0; kk < 4; ++kk) pk[kk] = (__bf16)wreg[kk][j];   \
        *(v4bf*)&sB[0][(n4 + j) * kLds + kq * 4] = pk;                                   \
    }                                                                                    \
    wait_async();                                                                        \
    __syncthreads();                                                                     \
    int cb = 0;                                                                          \
    for (int k0 = 0; k0 < (KDIM); k0 += 32) {                                            \
        const bool has_next = (k0 + 32) < (KDIM);                                        \
        if (has_next) {                                                                  \
            async_copy_32B(aSrc + k0 + 32, aDst[cb ^ 1]);                                \
            _Pragma("unroll") for (int kk = 0; kk < 4; ++kk) {                           \
                float4 q = *(const float4*)(bSrc + (size_t)(k0 + 32 + kk) * (LDB));      \
                wreg[kk][0] = q.x; wreg[kk][1] = q.y;                                    \
                wreg[kk][2] = q.z; wreg[kk][3] = q.w;                                    \
            }                                                                            \
        }                                                                                \
        v16bf af[2], bfm[4];                                                             \
        _Pragma("unroll") for (int fm = 0; fm < 2; ++fm)                                 \
            load_afrag(sA[cb], wm + fm * 16 + l16, lhi, af[fm]);                         \
        _Pragma("unroll") for (int fn = 0; fn < 4; ++fn)                                 \
            load_bfrag(sB[cb], wn + fn * 16 + l16, lhi, bfm[fn]);                        \
        _Pragma("unroll") for (int fm = 0; fm < 2; ++fm)                                 \
            _Pragma("unroll") for (int fn = 0; fn < 4; ++fn)                             \
                acc[fm][fn] = __builtin_amdgcn_wmma_f32_16x16x32_bf16(                   \
                    false, af[fm], false, bfm[fn], (short)0, acc[fm][fn], false, false); \
        if (has_next) {                                                                  \
            _Pragma("unroll") for (int j = 0; j < 4; ++j) {                              \
                v4bf pk;                                                                 \
                _Pragma("unroll") for (int kk = 0; kk < 4; ++kk)                         \
                    pk[kk] = (__bf16)wreg[kk][j];                                        \
                *(v4bf*)&sB[cb ^ 1][(n4 + j) * kLds + kq * 4] = pk;                      \
            }                                                                            \
            wait_async();                                                                \
        }                                                                                \
        __syncthreads();                                                                 \
        cb ^= 1;                                                                         \
    }

// GEMM1: h = gelu(x @ W1[e] + b1[e]) over compacted slots, output bf16.
__launch_bounds__(256)
__global__ void moe_gemm1(const __bf16* __restrict__ xb, const float* __restrict__ W1,
                          const float* __restrict__ b1, const int* __restrict__ counts,
                          const int* __restrict__ tlist, __bf16* __restrict__ hb) {
    const int e = blockIdx.x / kMT, mt = blockIdx.x % kMT;
    const int cnt = counts[e];
    const int m0 = mt * 128;
    if (m0 >= cnt) return;
    const int n0 = blockIdx.y * 128;
    const float* Wp = W1 + (size_t)e * kD * kH;

    MOE_GEMM_BODY(xb + (size_t)(arid >> 1) * kD, kD, kH, kD)

    // epilogue: bias + GELU -> bf16
#pragma unroll
    for (int fm = 0; fm < 2; ++fm) {
#pragma unroll
        for (int i = 0; i < 8; ++i) {
            int m = wm + fm * 16 + lhi * 8 + i;
            if (m0 + m < cnt) {
                const int rid = rowid[m];
#pragma unroll
                for (int fn = 0; fn < 4; ++fn) {
                    int n = wn + fn * 16 + l16;
                    float v = acc[fm][fn][i] + b1[e * kH + n0 + n];
                    hb[(size_t)rid * kH + n0 + n] = (__bf16)gelu_fast(v);
                }
            }
        }
    }
}

// GEMM2: y = h @ W2[e] + b2[e], output f32.
__launch_bounds__(256)
__global__ void moe_gemm2(const __bf16* __restrict__ hb, const float* __restrict__ W2,
                          const float* __restrict__ b2, const int* __restrict__ counts,
                          const int* __restrict__ tlist, float* __restrict__ yb) {
    const int e = blockIdx.x / kMT, mt = blockIdx.x % kMT;
    const int cnt = counts[e];
    const int m0 = mt * 128;
    if (m0 >= cnt) return;
    const int n0 = blockIdx.y * 128;
    const float* Wp = W2 + (size_t)e * kH * kO;

    MOE_GEMM_BODY(hb + (size_t)arid * kH, kH, kO, kH)

    // epilogue: bias -> f32
#pragma unroll
    for (int fm = 0; fm < 2; ++fm) {
#pragma unroll
        for (int i = 0; i < 8; ++i) {
            int m = wm + fm * 16 + lhi * 8 + i;
            if (m0 + m < cnt) {
                const int rid = rowid[m];
#pragma unroll
                for (int fn = 0; fn < 4; ++fn) {
                    int n = wn + fn * 16 + l16;
                    yb[(size_t)rid * kO + n0 + n] = acc[fm][fn][i] + b2[e * kO + n0 + n];
                }
            }
        }
    }
}

// Weighted combine of the two expert outputs + LayerNorm over O.
__global__ void moe_combine_ln(const float* __restrict__ yb, const float* __restrict__ wtok,
                               const float* __restrict__ gamma, const float* __restrict__ beta,
                               float* __restrict__ out) {
    const int t = blockIdx.x, tid = threadIdx.x;
    const float w0 = wtok[2 * t], w1 = wtok[2 * t + 1];
    const float* y0 = yb + (size_t)(2 * t) * kO;
    const float* y1 = y0 + kO;
    float v[4], sum = 0.f, sq = 0.f;
#pragma unroll
    for (int j = 0; j < 4; ++j) {
        int o = tid + j * 256;
        float z = w0 * y0[o] + w1 * y1[o];
        v[j] = z; sum += z; sq += z * z;
    }
#pragma unroll
    for (int off = 16; off; off >>= 1) {
        sum += __shfl_xor(sum, off, 32);
        sq  += __shfl_xor(sq,  off, 32);
    }
    __shared__ float rS[8], rQ[8];
    __shared__ float s_mu, s_rs;
    const int wid = tid >> 5, lane = tid & 31;
    if (lane == 0) { rS[wid] = sum; rQ[wid] = sq; }
    __syncthreads();
    if (tid == 0) {
        float S = 0.f, Q = 0.f;
#pragma unroll
        for (int i = 0; i < 8; ++i) { S += rS[i]; Q += rQ[i]; }
        float mu  = S / (float)kO;
        float var = Q / (float)kO - mu * mu;
        s_mu = mu; s_rs = rsqrtf(var + kEps);
    }
    __syncthreads();
    const float mu = s_mu, r = s_rs;
#pragma unroll
    for (int j = 0; j < 4; ++j) {
        int o = tid + j * 256;
        out[(size_t)t * kO + o] = (v[j] - mu) * r * gamma[o] + beta[o];
    }
}

extern "C" void kernel_launch(void* const* d_in, const int* in_sizes, int n_in,
                              void* d_out, int out_size, void* d_ws, size_t ws_size,
                              hipStream_t stream) {
    (void)in_sizes; (void)n_in; (void)out_size; (void)ws_size;
    const float* x     = (const float*)d_in[0];
    const float* Wr    = (const float*)d_in[1];
    const float* br    = (const float*)d_in[2];
    const float* W1    = (const float*)d_in[3];
    const float* b1    = (const float*)d_in[4];
    const float* W2    = (const float*)d_in[5];
    const float* b2    = (const float*)d_in[6];
    const float* gamma = (const float*)d_in[7];
    const float* beta  = (const float*)d_in[8];
    float* out = (float*)d_out;

    char* ws = (char*)d_ws;
    __bf16* xb   = (__bf16*)(ws + OFF_XB);
    __bf16* hb   = (__bf16*)(ws + OFF_HB);
    float*  yb   = (float*)(ws + OFF_YB);
    float*  wtok = (float*)(ws + OFF_WT);
    int*    cnts = (int*)(ws + OFF_CNT);
    int*    tlst = (int*)(ws + OFF_LST);

    moe_init<<<1, 32, 0, stream>>>(cnts);
    moe_cvt_x<<<(kT * kD) / 1024, 256, 0, stream>>>(x, xb);
    moe_router<<<kT / 8, 256, 0, stream>>>(x, Wr, br, cnts, tlst, wtok);
    moe_gemm1<<<dim3(kE * kMT, kH / 128), 256, 0, stream>>>(xb, W1, b1, cnts, tlst, hb);
    moe_gemm2<<<dim3(kE * kMT, kO / 128), 256, 0, stream>>>(hb, W2, b2, cnts, tlst, yb);
    moe_combine_ln<<<kT, 256, 0, stream>>>(yb, wtok, gamma, beta, out);
}
